// MMSGDTA_12154757448191
// MI455X (gfx1250) — compile-verified, matrix operand order checked by reading, not run on hardware
//
#include <hip/hip_runtime.h>
#include <hip/hip_bf16.h>

typedef __attribute__((ext_vector_type(16))) __bf16 v16bf;
typedef __attribute__((ext_vector_type(8)))  __bf16 v8bf;
typedef __attribute__((ext_vector_type(8)))  float  v8f;

// ---------- helpers ----------
static __device__ __forceinline__ unsigned fenc(float f) {
    unsigned u = __float_as_uint(f);
    return (u & 0x80000000u) ? ~u : (u | 0x80000000u);
}
static __device__ __forceinline__ float fdec(unsigned u) {
    return (u & 0x80000000u) ? __uint_as_float(u & 0x7fffffffu) : __uint_as_float(~u);
}
static __device__ __forceinline__ int kfrag(int j, int half) {
    return ((j < 4) ? (2 * j) : (16 + 2 * (j - 4))) + half * 8;
}
// WMMA 16-bit fragment = two contiguous 8-elem K-runs -> two b128 LDS loads.
static __device__ __forceinline__ v16bf frag_from_lds(const __bf16* p) {
    const v8bf lo = *(const v8bf*)(p);        // k = half*8 .. +7
    const v8bf hi = *(const v8bf*)(p + 16);   // k = 16+half*8 .. +7
    return __builtin_shufflevector(lo, hi, 0, 1, 2, 3, 4, 5, 6, 7,
                                   8, 9, 10, 11, 12, 13, 14, 15);
}

__global__ void k_fill_f(float* p, float v, int n) {
    int i = blockIdx.x * blockDim.x + threadIdx.x;
    if (i < n) p[i] = v;
}
__global__ void k_fill_u(unsigned* p, unsigned v, int n) {
    int i = blockIdx.x * blockDim.x + threadIdx.x;
    if (i < n) p[i] = v;
}

// ---------- FAST GEMM: LDS-staged, M%64==0, N%128==0, K%32==0 ----------
// 8 waves: block computes 64(M) x 128(N); wave (wx,wy) does 16 x 64 strip.
__global__ __launch_bounds__(256)
void k_gemm_wmma_fast(const float* __restrict__ A, const float* __restrict__ B,
                      const float* __restrict__ bias, float* __restrict__ C,
                      int M, int N, int K, int relu)
{
    __shared__ __bf16 Asl[64 * 40];    // 64 rows x 32k, padded to 40
    __shared__ __bf16 Bsl[128 * 40];   // 128 n-rows x 32k (transposed), padded
    const int tid  = threadIdx.x;
    const int w    = tid >> 5;
    const int lane = tid & 31;
    const int wx   = w & 3;            // m-tile within block
    const int wy   = w >> 2;           // 64-wide n strip
    const int half = lane >> 4;
    const int mn   = lane & 15;
    const int tm0  = blockIdx.y * 64;
    const int tn0  = blockIdx.x * 128;
    const v8f z = {0.f, 0.f, 0.f, 0.f, 0.f, 0.f, 0.f, 0.f};
    v8f acc[4] = {z, z, z, z};
    for (int k0 = 0; k0 < K; k0 += 32) {
        // stage A tile: 64x32 f32 -> bf16 (1024 float2, 4/thread, coalesced)
#pragma unroll
        for (int q = 0; q < 4; ++q) {
            const int p   = tid + 256 * q;
            const int row = p >> 4;
            const int kk2 = p & 15;
            const float2 f = *(const float2*)(A + (size_t)(tm0 + row) * K + k0 + 2 * kk2);
            Asl[row * 40 + 2 * kk2]     = (__bf16)f.x;
            Asl[row * 40 + 2 * kk2 + 1] = (__bf16)f.y;
        }
        // stage B tile transposed: 32k x 128n f32 -> bf16[n][k] (2048 float2, 8/thread)
#pragma unroll
        for (int q = 0; q < 8; ++q) {
            const int p  = tid + 256 * q;
            const int kr = p >> 6;
            const int n2 = p & 63;
            const float2 f = *(const float2*)(B + (size_t)(k0 + kr) * N + tn0 + 2 * n2);
            Bsl[(2 * n2) * 40 + kr]     = (__bf16)f.x;
            Bsl[(2 * n2 + 1) * 40 + kr] = (__bf16)f.y;
        }
        __syncthreads();
        const v16bf af = frag_from_lds(&Asl[(wx * 16 + mn) * 40 + half * 8]);
#pragma unroll
        for (int u = 0; u < 4; ++u) {
            const v16bf bfv = frag_from_lds(&Bsl[(wy * 64 + u * 16 + mn) * 40 + half * 8]);
            acc[u] = __builtin_amdgcn_wmma_f32_16x16x32_bf16(false, af, false, bfv,
                                                             (short)0, acc[u], false, false);
        }
        __syncthreads();
    }
#pragma unroll
    for (int u = 0; u < 4; ++u) {
        const int n = tn0 + wy * 64 + u * 16 + mn;
        const float bv = bias ? bias[n] : 0.f;
#pragma unroll
        for (int r = 0; r < 8; ++r) {
            const int mr = tm0 + wx * 16 + half * 8 + r;
            float v = acc[u][r] + bv;
            if (relu) v = fmaxf(v, 0.f);
            C[(size_t)mr * N + n] = v;
        }
    }
}

// ---------- generic GEMM (ragged K; M%16==0, N%16==0), branch-free ----------
__global__ __launch_bounds__(32)
void k_gemm_wmma(const float* __restrict__ A, const float* __restrict__ B,
                 const float* __restrict__ bias, float* __restrict__ C,
                 int M, int N, int K, int relu)
{
    const int tn = blockIdx.x * 16;
    const int tm = blockIdx.y * 16;
    const int lane = threadIdx.x & 31;
    const int half = lane >> 4;
    const int mn = lane & 15;
    const int m = tm + mn;
    const int n = tn + mn;
    v8f acc = {0.f, 0.f, 0.f, 0.f, 0.f, 0.f, 0.f, 0.f};
    for (int k0 = 0; k0 < K; k0 += 32) {
        v16bf af, bfv;
#pragma unroll
        for (int j = 0; j < 8; ++j) {
            const int kb = k0 + kfrag(j, half);
#pragma unroll
            for (int s2 = 0; s2 < 2; ++s2) {
                const int k = kb + s2;
                const int kc = k < K ? k : (K - 1);
                const float av = A[(size_t)m * K + kc];
                const float bv = B[(size_t)kc * N + n];
                af[2 * j + s2]  = (__bf16)(k < K ? av : 0.f);
                bfv[2 * j + s2] = (__bf16)(k < K ? bv : 0.f);
            }
        }
        acc = __builtin_amdgcn_wmma_f32_16x16x32_bf16(false, af, false, bfv,
                                                      (short)0, acc, false, false);
    }
#pragma unroll
    for (int r = 0; r < 8; ++r) {
        const int mr = tm + half * 8 + r;
        float v = acc[r];
        if (bias) v += bias[n];
        if (relu) v = fmaxf(v, 0.f);
        C[(size_t)mr * N + n] = v;
    }
}

// ---------- conv1d (k=3, VALID): im2col tile staged once in LDS ----------
// X: [B,Cin,Lin], W: [96, Cin*3], Y: [B,96,Lout], bias+relu.
// 6 waves (192 thr): block = all 96 out-ch x 64 positions; X fetched once.
__global__ __launch_bounds__(192)
void k_conv1d_wmma(const float* __restrict__ X, const float* __restrict__ W,
                   const float* __restrict__ bias, float* __restrict__ Y,
                   int Cin, int Lin)
{
    __shared__ __bf16 Xsl[64 * 392];          // [t][k], rows padded to K+8 (max 392)
    const int Lout = Lin - 2;
    const int K = Cin * 3;                    // 384 or 288
    const int LDB = K + 8;                    // row stride: multiple of 8 elems (16B)
    const int b  = blockIdx.y;
    const int tt = blockIdx.x * 64;
    const int tid  = threadIdx.x;
    const int w    = tid >> 5;                // o-tile 0..5
    const int lane = tid & 31;
    const int half = lane >> 4;
    const int mn   = lane & 15;
    const float* Xb = X + (size_t)b * Cin * Lin;
    // stage im2col tile: thread owns t-column (tid&63) and tap g (tid>>6)
    {
        const int t = tid & 63;
        const int g = tid >> 6;               // 0..2
        for (int ci = 0; ci < Cin; ++ci) {
            const int tg = tt + t + g;
            const int tc = tg < Lin ? tg : (Lin - 1);
            const float v = Xb[(size_t)ci * Lin + tc];
            Xsl[t * LDB + ci * 3 + g] = (__bf16)(tg < Lin ? v : 0.f);
        }
    }
    __syncthreads();
    const float* Wrow = W + (size_t)(w * 16 + mn) * K;
    const v8f z = {0.f, 0.f, 0.f, 0.f, 0.f, 0.f, 0.f, 0.f};
    v8f acc[4] = {z, z, z, z};
    for (int k0 = 0; k0 < K; k0 += 32) {
        v16bf af;
#pragma unroll
        for (int j = 0; j < 8; ++j) {
            const float2 av = *(const float2*)(Wrow + k0 + kfrag(j, half));
            af[2 * j]     = (__bf16)av.x;
            af[2 * j + 1] = (__bf16)av.y;
        }
#pragma unroll
        for (int u = 0; u < 4; ++u) {
            const v16bf bfv = frag_from_lds(&Xsl[(u * 16 + mn) * LDB + k0 + half * 8]);
            acc[u] = __builtin_amdgcn_wmma_f32_16x16x32_bf16(false, af, false, bfv,
                                                             (short)0, acc[u], false, false);
        }
    }
#pragma unroll
    for (int u = 0; u < 4; ++u) {
        const int t = tt + u * 16 + mn;
        if (t < Lout) {
#pragma unroll
            for (int r = 0; r < 8; ++r) {
                const int o = w * 16 + half * 8 + r;
                Y[((size_t)b * 96 + o) * Lout + t] = fmaxf(acc[u][r] + bias[o], 0.f);
            }
        }
    }
}

// ---------- GAT pieces ----------
__global__ void k_gat_att(const float* __restrict__ h, const float* __restrict__ asw,
                          const float* __restrict__ adw, float* __restrict__ osrc,
                          float* __restrict__ odst, int N)
{
    int i = blockIdx.x * blockDim.x + threadIdx.x;
    if (i >= N * 4) return;
    const int hh = i & 3;
    const float* hp  = h   + (size_t)(i >> 2) * 512 + hh * 128;
    const float* asp = asw + hh * 128;
    const float* adp = adw + hh * 128;
    float s = 0.f, d = 0.f;
    for (int c = 0; c < 128; ++c) { const float v = hp[c]; s += v * asp[c]; d += v * adp[c]; }
    osrc[i] = s; odst[i] = d;
}

__global__ void k_gat_edge_max(const int* __restrict__ ei, int E, int N,
                               const float* __restrict__ asrc, const float* __restrict__ adst,
                               unsigned* __restrict__ mmax)
{
    int i = blockIdx.x * blockDim.x + threadIdx.x;
    const int tot = (E + N) * 4;
    if (i >= tot) return;
    const int hh = i & 3, e = i >> 2;
    int s, d;
    if (e < E) { s = ei[e]; d = ei[E + e]; } else { s = d = e - E; }
    float v = asrc[s * 4 + hh] + adst[d * 4 + hh];
    v = v > 0.f ? v : 0.2f * v;
    atomicMax(&mmax[d * 4 + hh], fenc(v));
}

__global__ void k_gat_edge_acc(const int* __restrict__ ei, int E, int N,
                               const float* __restrict__ asrc, const float* __restrict__ adst,
                               const unsigned* __restrict__ mmax, float* __restrict__ denom,
                               const float* __restrict__ h, float* __restrict__ accum)
{
    int i = blockIdx.x * blockDim.x + threadIdx.x;
    const int tot = (E + N) * 4;
    if (i >= tot) return;
    const int hh = i & 3, e = i >> 2;
    int s, d;
    if (e < E) { s = ei[e]; d = ei[E + e]; } else { s = d = e - E; }
    float v = asrc[s * 4 + hh] + adst[d * 4 + hh];
    v = v > 0.f ? v : 0.2f * v;
    const float al = __expf(v - fdec(mmax[d * 4 + hh]));
    atomicAdd(&denom[d * 4 + hh], al);
    const float* hv = h     + ((size_t)s * 4 + hh) * 128;
    float*       ov = accum + ((size_t)d * 4 + hh) * 128;
    for (int c = 0; c < 128; ++c) atomicAdd(&ov[c], al * hv[c]);
}

__global__ void k_gat_final(const float* __restrict__ accum, const float* __restrict__ denom,
                            const float* __restrict__ bias, float* __restrict__ out,
                            int N, int relu)
{
    int i = blockIdx.x * blockDim.x + threadIdx.x;
    if (i >= N * 128) return;
    const int n = i >> 7, c = i & 127;
    float s = 0.f;
    for (int hh = 0; hh < 4; ++hh)
        s += accum[((size_t)n * 4 + hh) * 128 + c] / denom[n * 4 + hh];
    float v = 0.25f * s + bias[c];
    if (relu) v = fmaxf(v, 0.f);
    out[i] = v;
}

// ---------- GCN pieces ----------
__global__ void k_gcn_deg(const int* __restrict__ ei, const float* __restrict__ w,
                          int E, int N, float* __restrict__ deg)
{
    int i = blockIdx.x * blockDim.x + threadIdx.x;
    if (i >= E + N) return;
    if (i < E) atomicAdd(&deg[ei[E + i]], w[i]);
    else       atomicAdd(&deg[i - E], 1.0f);
}

__global__ void k_gcn_acc(const int* __restrict__ ei, const float* __restrict__ w,
                          int E, int N, const float* __restrict__ deg,
                          const float* __restrict__ h, float* __restrict__ out)
{
    int e = blockIdx.x * blockDim.x + threadIdx.x;
    if (e >= E + N) return;
    int s, d; float wt;
    if (e < E) { s = ei[e]; d = ei[E + e]; wt = w[e]; } else { s = d = e - E; wt = 1.f; }
    const float ds = deg[s], dd = deg[d];
    const float norm = (ds > 0.f ? rsqrtf(ds) : 0.f) * wt * (dd > 0.f ? rsqrtf(dd) : 0.f);
    const float* hs = h + (size_t)s * 128;
    float* od = out + (size_t)d * 128;
    for (int c = 0; c < 128; ++c) atomicAdd(&od[c], norm * hs[c]);
}

__global__ void k_bias_act(float* __restrict__ x, const float* __restrict__ b,
                           int total, int C, int relu)
{
    int i = blockIdx.x * blockDim.x + threadIdx.x;
    if (i >= total) return;
    float v = x[i] + b[i % C];
    if (relu) v = fmaxf(v, 0.f);
    x[i] = v;
}

__global__ void k_gate_fuse(float* __restrict__ h, const float* __restrict__ hc,
                            const float* __restrict__ t1, const float* __restrict__ t2,
                            const float* __restrict__ pbias, int total)
{
    int i = blockIdx.x * blockDim.x + threadIdx.x;
    if (i >= total) return;
    const int c = i & 127;
    const float z = 1.f / (1.f + __expf(-(t1[i] + t2[i] + pbias[c])));
    h[i] = z * hc[i] + (1.f - z) * h[i];
}

// ---------- pooling ----------
__global__ void k_mean_pool(const float* __restrict__ x, const int* __restrict__ batch,
                            float* __restrict__ pool, float* __restrict__ cnt, int N)
{
    int i = blockIdx.x * blockDim.x + threadIdx.x;
    if (i >= N * 128) return;
    const int n = i >> 7, c = i & 127;
    const int b = batch[n];
    atomicAdd(&pool[b * 128 + c], x[i]);
    if (c == 0) atomicAdd(&cnt[b], 1.f);
}
__global__ void k_mean_div(float* __restrict__ pool, const float* __restrict__ cnt, int n)
{
    int i = blockIdx.x * blockDim.x + threadIdx.x;
    if (i < n) pool[i] /= cnt[i >> 7];
}
__global__ void k_max_pool(const float* __restrict__ x, const int* __restrict__ batch,
                           unsigned* __restrict__ pool, int N)
{
    int i = blockIdx.x * blockDim.x + threadIdx.x;
    if (i >= N * 128) return;
    atomicMax(&pool[batch[i >> 7] * 128 + (i & 127)], fenc(x[i]));
}
__global__ void k_dec_pool(const unsigned* __restrict__ pu, float* __restrict__ pf, int n)
{
    int i = blockIdx.x * blockDim.x + threadIdx.x;
    if (i < n) pf[i] = fdec(pu[i]);
}

// ---------- sequence branch ----------
__global__ void k_embed_gather(const int* __restrict__ seq, const float* __restrict__ emb,
                               float* __restrict__ s, int total)
{
    int i = blockIdx.x * blockDim.x + threadIdx.x;
    if (i >= total) return;
    const int b = i / (128 * 1000);
    const int rem = i - b * 128 * 1000;
    const int e = rem / 1000;
    const int l = rem - e * 1000;
    s[i] = emb[(size_t)seq[b * 1000 + l] * 128 + e];
}

__global__ void k_maxpool_len(const float* __restrict__ y, float* __restrict__ feats,
                              int L, int colOff)
{
    int i = blockIdx.x * blockDim.x + threadIdx.x;
    if (i >= 64 * 96) return;
    const int b = i / 96, o = i - b * 96;
    const float* p = y + ((size_t)b * 96 + o) * L;
    float m = -3.0e38f;
    for (int t = 0; t < L; ++t) m = fmaxf(m, p[t]);
    feats[(size_t)b * 288 + colOff + o] = m;
}

// ---------- attention fusion + head ----------
__global__ void k_attn_proj(const float* __restrict__ v, const float* __restrict__ w1,
                            const float* __restrict__ b1, const float* __restrict__ w2,
                            float* __restrict__ scores, int col)
{
    int b = blockIdx.x * blockDim.x + threadIdx.x;
    if (b >= 64) return;
    const float* vb = v + b * 128;
    float s = 0.f;
    for (int j = 0; j < 64; ++j) {
        float t = b1[j];
        for (int c = 0; c < 128; ++c) t += vb[c] * w1[c * 64 + j];
        s += tanhf(t) * w2[j];
    }
    scores[b * 3 + col] = s;
}

__global__ void k_attn_combine(const float* __restrict__ sc, const float* __restrict__ x,
                               const float* __restrict__ xt, const float* __restrict__ st,
                               float* __restrict__ emb)
{
    int i = blockIdx.x * blockDim.x + threadIdx.x;
    if (i >= 64 * 128) return;
    const int b = i >> 7;
    const float s0 = sc[b * 3], s1 = sc[b * 3 + 1], s2 = sc[b * 3 + 2];
    const float m = fmaxf(s0, fmaxf(s1, s2));
    const float e0 = __expf(s0 - m), e1 = __expf(s1 - m), e2 = __expf(s2 - m);
    const float inv = 1.f / (e0 + e1 + e2);
    emb[i] = (e0 * x[i] + e1 * xt[i] + e2 * st[i]) * inv;
}

__global__ void k_final_fc(const float* __restrict__ f2, const float* __restrict__ w,
                           const float* __restrict__ b, float* __restrict__ out)
{
    int i = blockIdx.x * blockDim.x + threadIdx.x;
    if (i >= 64) return;
    float s = b[0];
    const float* p = f2 + (size_t)i * 512;
    for (int c = 0; c < 512; ++c) s += p[c] * w[c];
    out[i] = s;
}

// =====================================================================
static inline size_t al256(size_t x) { return (x + 255) & ~(size_t)255; }
static inline int cdiv(int a, int b) { return (a + b - 1) / b; }

static void gemm(const float* A, const float* B, const float* bias, float* C,
                 int M, int N, int K, int relu, hipStream_t stream)
{
    if ((M & 63) == 0 && (N & 127) == 0 && (K & 31) == 0)
        k_gemm_wmma_fast<<<dim3(N / 128, M / 64), 256, 0, stream>>>(A, B, bias, C, M, N, K, relu);
    else
        k_gemm_wmma<<<dim3(cdiv(N, 16), cdiv(M, 16)), 32, 0, stream>>>(A, B, bias, C, M, N, K, relu);
}

#define FILLF(p, v, n)  k_fill_f<<<cdiv((n),256),256,0,stream>>>((p),(v),(n))
#define FILLU(p, v, n)  k_fill_u<<<cdiv((n),256),256,0,stream>>>((p),(v),(n))

extern "C" void kernel_launch(void* const* d_in, const int* in_sizes, int n_in,
                              void* d_out, int out_size, void* d_ws, size_t ws_size,
                              hipStream_t stream)
{
    // ---- inputs ----
    const float* mol_x   = (const float*)d_in[0];
    const int*   mol_ei  = (const int*)d_in[1];
    const int*   mol_b   = (const int*)d_in[2];
    const float* pro_x   = (const float*)d_in[3];
    const int*   pro_ei  = (const int*)d_in[4];
    const float* pro_ew  = (const float*)d_in[5];
    const int*   pro_b   = (const int*)d_in[6];
    const int*   seq     = (const int*)d_in[7];
    const int Nm = in_sizes[0] / 78;
    const int Em = in_sizes[1] / 2;
    const int Np = in_sizes[3] / 32;
    const int Ep = in_sizes[5];

    // ---- params (jax pytree leaf order: sorted dict keys) ----
    const float* att_st_b  = (const float*)d_in[8];
    const float* att_st_w  = (const float*)d_in[9];
    const float* att_st_l2 = (const float*)d_in[10];
    const float* att_x_b   = (const float*)d_in[11];
    const float* att_x_w   = (const float*)d_in[12];
    const float* att_x_l2  = (const float*)d_in[13];
    const float* att_xt_b  = (const float*)d_in[14];
    const float* att_xt_w  = (const float*)d_in[15];
    const float* att_xt_l2 = (const float*)d_in[16];
    const float* fc1_b = (const float*)d_in[17];
    const float* fc1_w = (const float*)d_in[18];
    const float* fc2_b = (const float*)d_in[19];
    const float* fc2_w = (const float*)d_in[20];
    const float* fc3_b = (const float*)d_in[21];
    const float* fc3_w = (const float*)d_in[22];
    const float* mconv_ad[3] = {(const float*)d_in[23], (const float*)d_in[27], (const float*)d_in[31]};
    const float* mconv_as[3] = {(const float*)d_in[24], (const float*)d_in[28], (const float*)d_in[32]};
    const float* mconv_b [3] = {(const float*)d_in[25], (const float*)d_in[29], (const float*)d_in[33]};
    const float* mconv_w [3] = {(const float*)d_in[26], (const float*)d_in[30], (const float*)d_in[34]};
    const float* mgfc1_b = (const float*)d_in[35];
    const float* mgfc1_w = (const float*)d_in[36];
    const float* mgfc2_b = (const float*)d_in[37];
    const float* mgfc2_w = (const float*)d_in[38];
    const float* mlin1_b = (const float*)d_in[39];
    const float* mlin1_w = (const float*)d_in[40];
    const float* mlin2_b = (const float*)d_in[41];
    const float* mlin2_w = (const float*)d_in[42];
    const float* pbias   = (const float*)d_in[43];
    const float* pgat_ad[2] = {(const float*)d_in[44], (const float*)d_in[48]};
    const float* pgat_as[2] = {(const float*)d_in[45], (const float*)d_in[49]};
    const float* pgat_b [2] = {(const float*)d_in[46], (const float*)d_in[50]};
    const float* pgat_w [2] = {(const float*)d_in[47], (const float*)d_in[51]};
    const float* pgcn_b  = (const float*)d_in[52];
    const float* pgcn_w  = (const float*)d_in[53];
    const float* pgfc1_b = (const float*)d_in[54];
    const float* pgfc1_w = (const float*)d_in[55];
    const float* pgfc2_b = (const float*)d_in[56];
    const float* pgfc2_w = (const float*)d_in[57];
    const float* pfc1_b  = (const float*)d_in[58];
    const float* pfc1_w  = (const float*)d_in[59];
    const float* pfc2_b  = (const float*)d_in[60];
    const float* pfc2_w  = (const float*)d_in[61];
    const float* cb_b[6] = {(const float*)d_in[62], (const float*)d_in[64], (const float*)d_in[66],
                            (const float*)d_in[68], (const float*)d_in[70], (const float*)d_in[72]};
    const float* cb_w[6] = {(const float*)d_in[63], (const float*)d_in[65], (const float*)d_in[67],
                            (const float*)d_in[69], (const float*)d_in[71], (const float*)d_in[73]};
    const float* sembed  = (const float*)d_in[74];
    const float* sout_b  = (const float*)d_in[75];
    const float* sout_w  = (const float*)d_in[76];

    // ---- workspace carve ----
    char* ws = (char*)d_ws;
    size_t off = 0;
    auto alloc = [&](size_t elems) -> float* {
        float* p = (float*)(ws + off); off = al256(off + elems * 4); return p;
    };
    float* molx  = alloc(64 * 128);
    float* xtemb = alloc(64 * 128);
    float* stemb = alloc(64 * 128);
    float* feats = alloc(64 * 288);
    float* scores = alloc(64 * 3);
    float* emb   = alloc(64 * 128);
    float* f1    = alloc(64 * 1024);
    float* f2    = alloc(64 * 512);
    float* pool  = alloc(64 * 128);
    float* cnt   = alloc(64);
    unsigned* poolu = (unsigned*)alloc(64 * 128);
    float* hid   = alloc(64 * 1024);
    const size_t phase0 = off;
    const unsigned NEG = 0x00000041u; // encoded large-negative floor for float atomicMax

    // =================== MOL GRAPH BRANCH ===================
    {
        off = phase0;
        float* x1 = alloc((size_t)Nm * 128);
        float* x2 = alloc((size_t)Nm * 128);
        float* gh = alloc((size_t)Nm * 512);
        float* as = alloc(Nm * 4);
        float* ad = alloc(Nm * 4);
        unsigned* mm = (unsigned*)alloc(Nm * 4);
        float* dn = alloc(Nm * 4);
        float* ac = alloc((size_t)Nm * 512);

        gemm(mol_x, mlin1_w, mlin1_b, x1, Nm, 128, 78, 1, stream);
        gemm(x1, mlin2_w, mlin2_b, x2, Nm, 128, 128, 0, stream);
        float* xcur = x2; float* xnxt = x1;
        const int ET4 = (Em + Nm) * 4;
        for (int li = 0; li < 3; ++li) {
            gemm(xcur, mconv_w[li], nullptr, gh, Nm, 512, 128, 0, stream);
            k_gat_att<<<cdiv(Nm * 4, 256), 256, 0, stream>>>(gh, mconv_as[li], mconv_ad[li], as, ad, Nm);
            FILLU(mm, NEG, Nm * 4);
            FILLF(dn, 0.f, Nm * 4);
            FILLF(ac, 0.f, Nm * 512);
            k_gat_edge_max<<<cdiv(ET4, 256), 256, 0, stream>>>(mol_ei, Em, Nm, as, ad, mm);
            k_gat_edge_acc<<<cdiv(ET4, 256), 256, 0, stream>>>(mol_ei, Em, Nm, as, ad, mm, dn, gh, ac);
            k_gat_final<<<cdiv(Nm * 128, 256), 256, 0, stream>>>(ac, dn, mconv_b[li], xnxt, Nm, 0);
            float* t = xcur; xcur = xnxt; xnxt = t;
        }
        FILLF(pool, 0.f, 64 * 128);
        FILLF(cnt, 0.f, 64);
        k_mean_pool<<<cdiv(Nm * 128, 256), 256, 0, stream>>>(xcur, mol_b, pool, cnt, Nm);
        k_mean_div<<<cdiv(64 * 128, 256), 256, 0, stream>>>(pool, cnt, 64 * 128);
        gemm(pool, mgfc1_w, mgfc1_b, hid, 64, 1024, 128, 1, stream);
        gemm(hid, mgfc2_w, mgfc2_b, molx, 64, 128, 1024, 0, stream);
    }

    // =================== PROTEIN GRAPH BRANCH ===================
    {
        off = phase0;
        float* h   = alloc((size_t)Np * 128);
        float* hc  = alloc((size_t)Np * 128);
        float* t1  = alloc((size_t)Np * 128);
        float* t2  = alloc((size_t)Np * 128);
        float* gh  = alloc((size_t)Np * 512);
        float* ac  = alloc((size_t)Np * 512);
        float* as  = alloc(Np * 4);
        float* ad  = alloc(Np * 4);
        unsigned* mm = (unsigned*)alloc(Np * 4);
        float* dn  = alloc(Np * 4);
        float* deg = alloc(Np);

        gemm(pro_x, pgcn_w, nullptr, t1, Np, 128, 32, 0, stream);
        FILLF(h, 0.f, Np * 128);
        FILLF(deg, 0.f, Np);
        k_gcn_deg<<<cdiv(Ep + Np, 256), 256, 0, stream>>>(pro_ei, pro_ew, Ep, Np, deg);
        k_gcn_acc<<<cdiv(Ep + Np, 256), 256, 0, stream>>>(pro_ei, pro_ew, Ep, Np, deg, t1, h);
        k_bias_act<<<cdiv(Np * 128, 256), 256, 0, stream>>>(h, pgcn_b, Np * 128, 128, 1);

        const int ET4 = (Ep + Np) * 4;
        for (int li = 0; li < 2; ++li) {
            gemm(h, pgat_w[li], nullptr, gh, Np, 512, 128, 0, stream);
            k_gat_att<<<cdiv(Np * 4, 256), 256, 0, stream>>>(gh, pgat_as[li], pgat_ad[li], as, ad, Np);
            FILLU(mm, NEG, Np * 4);
            FILLF(dn, 0.f, Np * 4);
            FILLF(ac, 0.f, Np * 512);
            k_gat_edge_max<<<cdiv(ET4, 256), 256, 0, stream>>>(pro_ei, Ep, Np, as, ad, mm);
            k_gat_edge_acc<<<cdiv(ET4, 256), 256, 0, stream>>>(pro_ei, Ep, Np, as, ad, mm, dn, gh, ac);
            k_gat_final<<<cdiv(Np * 128, 256), 256, 0, stream>>>(ac, dn, pgat_b[li], hc, Np, li == 0 ? 1 : 0);
            gemm(hc, pfc1_w, pfc1_b, t1, Np, 128, 128, 0, stream);
            gemm(h,  pfc2_w, pfc2_b, t2, Np, 128, 128, 0, stream);
            k_gate_fuse<<<cdiv(Np * 128, 256), 256, 0, stream>>>(h, hc, t1, t2, pbias, Np * 128);
        }
        FILLU(poolu, NEG, 64 * 128);
        k_max_pool<<<cdiv(Np * 128, 256), 256, 0, stream>>>(h, pro_b, poolu, Np);
        k_dec_pool<<<cdiv(64 * 128, 256), 256, 0, stream>>>(poolu, pool, 64 * 128);
        gemm(pool, pgfc1_w, pgfc1_b, hid, 64, 1024, 128, 1, stream);
        gemm(hid, pgfc2_w, pgfc2_b, xtemb, 64, 128, 1024, 0, stream);
    }

    // =================== SEQUENCE BRANCH ===================
    {
        off = phase0;
        float* semb = alloc((size_t)64 * 128 * 1000);
        float* sc1  = alloc((size_t)64 * 96 * 998);
        float* sc2  = alloc((size_t)64 * 96 * 998);
        k_embed_gather<<<cdiv(64 * 128 * 1000, 256), 256, 0, stream>>>(seq, sembed, semb, 64 * 128 * 1000);
        dim3 cg0(cdiv(998, 64), 64), cg1(cdiv(996, 64), 64), cg2(cdiv(994, 64), 64);
        // block 0
        k_conv1d_wmma<<<cg0, 192, 0, stream>>>(semb, cb_w[0], cb_b[0], sc1, 128, 1000);
        k_maxpool_len<<<cdiv(64 * 96, 256), 256, 0, stream>>>(sc1, feats, 998, 0);
        // block 1
        k_conv1d_wmma<<<cg0, 192, 0, stream>>>(semb, cb_w[1], cb_b[1], sc1, 128, 1000);
        k_conv1d_wmma<<<cg1, 192, 0, stream>>>(sc1, cb_w[2], cb_b[2], sc2, 96, 998);
        k_maxpool_len<<<cdiv(64 * 96, 256), 256, 0, stream>>>(sc2, feats, 996, 96);
        // block 2
        k_conv1d_wmma<<<cg0, 192, 0, stream>>>(semb, cb_w[3], cb_b[3], sc1, 128, 1000);
        k_conv1d_wmma<<<cg1, 192, 0, stream>>>(sc1, cb_w[4], cb_b[4], sc2, 96, 998);
        k_conv1d_wmma<<<cg2, 192, 0, stream>>>(sc2, cb_w[5], cb_b[5], sc1, 96, 996);
        k_maxpool_len<<<cdiv(64 * 96, 256), 256, 0, stream>>>(sc1, feats, 994, 192);
        gemm(feats, sout_w, sout_b, stemb, 64, 128, 288, 0, stream);
    }

    // =================== ATTENTION FUSION + HEAD ===================
    k_attn_proj<<<1, 64, 0, stream>>>(molx,  att_x_w,  att_x_b,  att_x_l2,  scores, 0);
    k_attn_proj<<<1, 64, 0, stream>>>(xtemb, att_xt_w, att_xt_b, att_xt_l2, scores, 1);
    k_attn_proj<<<1, 64, 0, stream>>>(stemb, att_st_w, att_st_b, att_st_l2, scores, 2);
    k_attn_combine<<<cdiv(64 * 128, 256), 256, 0, stream>>>(scores, molx, xtemb, stemb, emb);
    gemm(emb, fc1_w, fc1_b, f1, 64, 1024, 128, 1, stream);
    gemm(f1,  fc2_w, fc2_b, f2, 64, 512, 1024, 1, stream);
    k_final_fc<<<1, 64, 0, stream>>>(f2, fc3_w, fc3_b, (float*)d_out);

    (void)n_in; (void)out_size; (void)ws_size;
}
